// CPTTransformer_18468359372972
// MI455X (gfx1250) — matrix operand order repaired and checked by reading, not compiled, not run on hardware
//
#include <hip/hip_runtime.h>

typedef _Float16 v16h __attribute__((ext_vector_type(16)));
typedef float    v8f  __attribute__((ext_vector_type(8)));

union ABfrag { v16h h; uint4 u[2]; };

static __device__ __forceinline__ v8f wmma_f16(const ABfrag& a, const ABfrag& b, v8f c) {
    return __builtin_amdgcn_wmma_f32_16x16x32_f16(false, a.h, false, b.h, (short)0, c, false, false);
}

// ---------------- CDNA5 async global->LDS copy via inline ISA (ASYNCcnt path)
static __device__ __forceinline__ void cp_async16(void* lds, const void* g) {
    unsigned l = (unsigned)(uintptr_t)lds;   // flat->LDS: low 32 bits are the LDS offset
    asm volatile("global_load_async_to_lds_b128 %0, %1, off"
                 :
                 : "v"(l), "v"(g)
                 : "memory");
}

static __device__ __forceinline__ void wait_async_all() {
    asm volatile("s_wait_asynccnt 0" ::: "memory");
}

// load a 16x16 fragment (two b128 LDS reads): elements [off0, off0+8) and [off0+16, off0+24)
static __device__ __forceinline__ void load_frag(ABfrag& f, const _Float16* base) {
    f.u[0] = *(const uint4*)(base);
    f.u[1] = *(const uint4*)(base + 16);
}

// ---------------------------------------------------------------- convert f32 -> f16
__global__ __launch_bounds__(256) void cvt_f32_f16(const float* __restrict__ src,
                                                   _Float16* __restrict__ dst, size_t n) {
    size_t i = (size_t)blockIdx.x * blockDim.x + threadIdx.x;
    size_t stride = (size_t)gridDim.x * blockDim.x;
    for (; i < n; i += stride) dst[i] = (_Float16)src[i];
}

// ---------------------------------------------------------------- convert + transpose: WT[n][k] = (f16)W[k][n]
__global__ __launch_bounds__(256) void cvt_transpose_f16(const float* __restrict__ W,
                                                         _Float16* __restrict__ WT,
                                                         int K, int N) {
    __shared__ float tile[32][33];
    const int kt = blockIdx.y * 32, nt = blockIdx.x * 32;
    const int tx = threadIdx.x & 31, ty = threadIdx.x >> 5;   // ty 0..7
#pragma unroll
    for (int i = 0; i < 4; ++i)
        tile[ty + 8 * i][tx] = W[(size_t)(kt + ty + 8 * i) * N + nt + tx];
    __syncthreads();
#pragma unroll
    for (int i = 0; i < 4; ++i)
        WT[(size_t)(nt + ty + 8 * i) * K + kt + tx] = (_Float16)tile[tx][ty + 8 * i];
}

// ---------------------------------------------------------------- generic GEMM
// C = A(f16, MxK) @ B^T stored as BT(f16, NxK) + bias. 256 thr (8 waves), tile 128x128,
// k-step 32, double-buffered LDS, staging entirely via async global->LDS DMA,
// hand-pipelined fragment loads, coalesced epilogue through LDS.
// MODE: 0 = f16 row-major out, 1 = f32 row-major out, 2 = f16 transposed out ([n][m])
template <int MODE>
__global__ __launch_bounds__(256) void gemm_wmma(const _Float16* __restrict__ A,
                                                 const _Float16* __restrict__ BT,
                                                 const float* __restrict__ bias,
                                                 void* __restrict__ outp,
                                                 int M, int N, int K) {
    __shared__ __align__(16) char gsm[32768];
    _Float16* At = (_Float16*)gsm;               // [2][4096]  [row][k]
    _Float16* Bt = (_Float16*)(gsm + 16384);     // [2][4096]  [n][k]

    const int tid   = threadIdx.x;
    const int lane  = tid & 31;
    const int wv    = tid >> 5;        // wave id -> 16-row strip
    const int half_ = lane >> 4;
    const int lm    = lane & 15;

    const int n0 = blockIdx.x * 128;
    const int m0 = blockIdx.y * 128;

    auto stage = [&](int buf, int k0) {
        int r = tid >> 1, ks = (tid & 1) * 16;
        const _Float16* asrc = A + (size_t)(m0 + r) * K + k0 + ks;
        _Float16* adst = At + buf * 4096 + r * 32 + ks;
        cp_async16(adst, asrc);
        cp_async16(adst + 8, asrc + 8);
        const _Float16* bsrc = BT + (size_t)(n0 + r) * K + k0 + ks;
        _Float16* bdst = Bt + buf * 4096 + r * 32 + ks;
        cp_async16(bdst, bsrc);
        cp_async16(bdst + 8, bsrc + 8);
    };

    v8f acc[8] = {};

    stage(0, 0);
    wait_async_all();
    __syncthreads();

    for (int k0 = 0; k0 < K; k0 += 32) {
        const int cur = (k0 >> 5) & 1;
        if (k0 + 32 < K) stage(cur ^ 1, k0 + 32);   // DMA next tile under compute

        const _Float16* ab = At + cur * 4096 + (wv * 16 + lm) * 32 + half_ * 8;
        const _Float16* bb = Bt + cur * 4096 + lm * 32 + half_ * 8;

        ABfrag a;
        load_frag(a, ab);
        ABfrag bf[2];
        load_frag(bf[0], bb);
#pragma unroll
        for (int t = 0; t < 8; ++t) {
            if (t < 7) load_frag(bf[(t + 1) & 1], bb + (t + 1) * 16 * 32);
            acc[t] = wmma_f16(a, bf[t & 1], acc[t]);
        }

        wait_async_all();
        __syncthreads();
    }

    // ---- epilogue
    if constexpr (MODE == 2) {
        // transposed f16 out: 8 contiguous m per lane -> packed b128 direct
#pragma unroll
        for (int t = 0; t < 8; ++t) {
            int n = n0 + t * 16 + lm;
            float bbv = bias[n];
            union { uint4 u; _Float16 h[8]; } pk;
#pragma unroll
            for (int v = 0; v < 8; ++v) pk.h[v] = (_Float16)(acc[t][v] + bbv);
            int m = m0 + wv * 16 + 8 * half_;
            *(uint4*)((_Float16*)outp + (size_t)n * M + m) = pk.u;
        }
    } else if constexpr (MODE == 0) {
        // stage 128x128 f16 tile in LDS, then coalesced b128 stores
        _Float16* Ep = (_Float16*)gsm;
#pragma unroll
        for (int t = 0; t < 8; ++t) {
            float bbv = bias[n0 + t * 16 + lm];
#pragma unroll
            for (int v = 0; v < 8; ++v)
                Ep[(wv * 16 + v + 8 * half_) * 128 + t * 16 + lm] = (_Float16)(acc[t][v] + bbv);
        }
        __syncthreads();
        int r = tid >> 1, cs = (tid & 1) * 64;
        const uint4* srcp = (const uint4*)(Ep + r * 128 + cs);
        uint4* dstp = (uint4*)((_Float16*)outp + (size_t)(m0 + r) * N + n0 + cs);
#pragma unroll
        for (int i = 0; i < 8; ++i) dstp[i] = srcp[i];
    } else {
        // f32 out: two 64-row chunks through 32KB LDS, coalesced b128 stores
        float* Epf = (float*)gsm;
#pragma unroll
        for (int c = 0; c < 2; ++c) {
            if ((wv >> 2) == c) {
#pragma unroll
                for (int t = 0; t < 8; ++t) {
                    float bbv = bias[n0 + t * 16 + lm];
#pragma unroll
                    for (int v = 0; v < 8; ++v)
                        Epf[((wv & 3) * 16 + v + 8 * half_) * 128 + t * 16 + lm] =
                            acc[t][v] + bbv;
                }
            }
            __syncthreads();
            int r = tid >> 2, cs = (tid & 3) * 32;
            const uint4* srcp = (const uint4*)(Epf + r * 128 + cs);
            uint4* dstp = (uint4*)((float*)outp + (size_t)(m0 + 64 * c + r) * N + n0 + cs);
#pragma unroll
            for (int i = 0; i < 8; ++i) dstp[i] = srcp[i];
            __syncthreads();
        }
    }
}

// ---------------------------------------------------------------- windowed attention
// One block per (batch, window). V is pre-transposed: vT16[d][m], m = b*4096 + s.
__global__ __launch_bounds__(256) void attn_kernel(const _Float16* __restrict__ q16,
                                                   const _Float16* __restrict__ k16,
                                                   const _Float16* __restrict__ vT16,
                                                   _Float16* __restrict__ outw) {
    __shared__ __align__(16) _Float16 sm0[2][8192];  // phase1: Qt(4096)+Kt(4096); phase3: VtT(64x128)
    __shared__ __align__(16) _Float16 P[128 * 128];  // probabilities

    const int tid   = threadIdx.x;
    const int lane  = tid & 31;
    const int wv    = tid >> 5;
    const int half_ = lane >> 4;
    const int lm    = lane & 15;

    const int b = blockIdx.x / 63;
    const int w = blockIdx.x % 63;
    const size_t rowbase = (size_t)b * 4096 + (size_t)w * 64;

    // ---- phase 1: scores = Q @ K^T (double-buffered async staging)
    auto stageQK = [&](int buf, int k0) {
        int r = tid >> 1, ks = (tid & 1) * 16;
        const _Float16* qs = q16 + (rowbase + r) * 1024 + k0 + ks;
        _Float16* qd = &sm0[buf][r * 32 + ks];
        cp_async16(qd, qs);
        cp_async16(qd + 8, qs + 8);
        const _Float16* ksrc = k16 + (rowbase + r) * 1024 + k0 + ks;
        _Float16* kd = &sm0[buf][4096 + r * 32 + ks];
        cp_async16(kd, ksrc);
        cp_async16(kd + 8, ksrc + 8);
    };

    v8f acc[8] = {};
    stageQK(0, 0);
    wait_async_all();
    __syncthreads();

    for (int k0 = 0; k0 < 1024; k0 += 32) {
        const int cur = (k0 >> 5) & 1;
        if (k0 + 32 < 1024) stageQK(cur ^ 1, k0 + 32);

        const _Float16* ab = &sm0[cur][(wv * 16 + lm) * 32 + half_ * 8];
        const _Float16* bb = &sm0[cur][4096 + lm * 32 + half_ * 8];

        ABfrag a;
        load_frag(a, ab);
        ABfrag bf[2];
        load_frag(bf[0], bb);
#pragma unroll
        for (int t = 0; t < 8; ++t) {
            if (t < 7) load_frag(bf[(t + 1) & 1], bb + (t + 1) * 16 * 32);
            acc[t] = wmma_f16(a, bf[t & 1], acc[t]);
        }
        wait_async_all();
        __syncthreads();
    }

    // ---- phase 2: softmax over 128 keys (scale 1/sqrt(HEAD_DIM) = 0.125)
    const float scale = 0.125f;
#pragma unroll
    for (int v = 0; v < 8; ++v) {
        float rmax = -1e30f;
#pragma unroll
        for (int t = 0; t < 8; ++t) rmax = fmaxf(rmax, acc[t][v]);
#pragma unroll
        for (int m = 1; m < 16; m <<= 1) rmax = fmaxf(rmax, __shfl_xor(rmax, m, 32));
        float rsum = 0.f;
#pragma unroll
        for (int t = 0; t < 8; ++t) {
            float e = __expf((acc[t][v] - rmax) * scale);
            acc[t][v] = e;
            rsum += e;
        }
#pragma unroll
        for (int m = 1; m < 16; m <<= 1) rsum += __shfl_xor(rsum, m, 32);
        float rinv = 1.f / rsum;
        int prow = wv * 16 + v + 8 * half_;
#pragma unroll
        for (int t = 0; t < 8; ++t)
            P[prow * 128 + t * 16 + lm] = (_Float16)(acc[t][v] * rinv);
    }
    __syncthreads();

    // ---- phase 3: out = P @ V  (VtT[d][s] staged by async DMA from transposed V)
    auto stageV = [&](int buf, int n0) {
        int rr = tid >> 2, cs = (tid & 3) * 32;
        const _Float16* src = vT16 + (size_t)(n0 + rr) * 16384 + rowbase + cs;
        _Float16* dst = &sm0[buf][rr * 128 + cs];
        cp_async16(dst, src);
        cp_async16(dst + 8, src + 8);
        cp_async16(dst + 16, src + 16);
        cp_async16(dst + 24, src + 24);
    };

    stageV(0, 0);
    wait_async_all();
    __syncthreads();

    for (int n0 = 0; n0 < 1024; n0 += 64) {
        const int cur = (n0 >> 6) & 1;
        if (n0 + 64 < 1024) stageV(cur ^ 1, n0 + 64);
        const _Float16* VtT = &sm0[cur][0];
        const _Float16* arow = P + (wv * 16 + lm) * 128 + half_ * 8;

#pragma unroll
        for (int nt = 0; nt < 4; ++nt) {
            const _Float16* bb = VtT + (nt * 16 + lm) * 128 + half_ * 8;
            v8f o = {};
            ABfrag a[2], bf[2];
            load_frag(a[0], arow);
            load_frag(bf[0], bb);
#pragma unroll
            for (int kk = 0; kk < 4; ++kk) {
                if (kk < 3) {
                    load_frag(a[(kk + 1) & 1], arow + (kk + 1) * 32);
                    load_frag(bf[(kk + 1) & 1], bb + (kk + 1) * 32);
                }
                o = wmma_f16(a[kk & 1], bf[kk & 1], o);
            }
            size_t obase = (size_t)blockIdx.x * 128 * 1024;
#pragma unroll
            for (int v = 0; v < 8; ++v)
                outw[obase + (size_t)(wv * 16 + v + 8 * half_) * 1024 + n0 + nt * 16 + lm] =
                    (_Float16)o[v];
        }
        wait_async_all();
        __syncthreads();
    }
}

// ---------------------------------------------------------------- combine overlapping windows
__global__ __launch_bounds__(256) void combine_kernel(const _Float16* __restrict__ outw,
                                                      _Float16* __restrict__ y16) {
    const int row = blockIdx.x;           // 0..16383
    const int b = row >> 12;
    const int p = row & 4095;
    const int col = threadIdx.x * 4;
    const int wf = p >> 6;

    float s[4] = {0.f, 0.f, 0.f, 0.f};
    float cnt = 0.f;
#pragma unroll
    for (int c = 0; c < 2; ++c) {
        int ww = wf - 1 + c;
        if (ww >= 0 && ww <= 62) {
            size_t base = ((size_t)(b * 63 + ww) * 128 + (p - 64 * ww)) * 1024 + col;
#pragma unroll
            for (int i = 0; i < 4; ++i) s[i] += (float)outw[base + i];
            cnt += 1.f;
        }
    }
    float inv = 1.f / cnt;
    size_t ybase = (size_t)row * 1024 + col;
#pragma unroll
    for (int i = 0; i < 4; ++i) y16[ybase + i] = (_Float16)(s[i] * inv);
}

// ---------------------------------------------------------------- launcher
extern "C" void kernel_launch(void* const* d_in, const int* in_sizes, int n_in,
                              void* d_out, int out_size, void* d_ws, size_t ws_size,
                              hipStream_t stream) {
    const float* x  = (const float*)d_in[0];
    const float* Wq = (const float*)d_in[1];
    const float* bq = (const float*)d_in[2];
    const float* Wk = (const float*)d_in[3];
    const float* bk = (const float*)d_in[4];
    const float* Wv = (const float*)d_in[5];
    const float* bv = (const float*)d_in[6];
    const float* Wo = (const float*)d_in[7];
    const float* bo = (const float*)d_in[8];

    const size_t NX = (size_t)16384 * 1024;   // B*S*D elements
    const size_t NW = (size_t)1024 * 1024;

    _Float16* ws    = (_Float16*)d_ws;
    _Float16* x16   = ws;
    _Float16* q16   = x16 + NX;
    _Float16* k16   = q16 + NX;
    _Float16* vT16  = k16 + NX;               // V stored transposed [d][m]
    _Float16* wqT16 = vT16 + NX;              // all weights stored transposed [n][k]
    _Float16* wkT16 = wqT16 + NW;
    _Float16* wvT16 = wkT16 + NW;
    _Float16* woT16 = wvT16 + NW;
    _Float16* outw  = woT16 + NW;             // 252*128*1024 f16
    _Float16* y16   = x16;                    // alias: x16 dead after QKV GEMMs

    cvt_f32_f16<<<2048, 256, 0, stream>>>(x, x16, NX);

    dim3 tgrid(1024 / 32, 1024 / 32);
    cvt_transpose_f16<<<tgrid, 256, 0, stream>>>(Wq, wqT16, 1024, 1024);
    cvt_transpose_f16<<<tgrid, 256, 0, stream>>>(Wk, wkT16, 1024, 1024);
    cvt_transpose_f16<<<tgrid, 256, 0, stream>>>(Wv, wvT16, 1024, 1024);
    cvt_transpose_f16<<<tgrid, 256, 0, stream>>>(Wo, woT16, 1024, 1024);

    dim3 ggrid(1024 / 128, 16384 / 128);
    gemm_wmma<0><<<ggrid, 256, 0, stream>>>(x16, wqT16, bq, q16,  16384, 1024, 1024);
    gemm_wmma<0><<<ggrid, 256, 0, stream>>>(x16, wkT16, bk, k16,  16384, 1024, 1024);
    gemm_wmma<2><<<ggrid, 256, 0, stream>>>(x16, wvT16, bv, vT16, 16384, 1024, 1024);

    attn_kernel<<<4 * 63, 256, 0, stream>>>(q16, k16, vT16, outw);

    combine_kernel<<<16384, 256, 0, stream>>>(outw, y16);

    gemm_wmma<1><<<ggrid, 256, 0, stream>>>(y16, woT16, bo, (float*)d_out, 16384, 1024, 1024);
}
